// UnrollingModel_35132832481783
// MI455X (gfx1250) — compile-verified, matrix operand order checked by reading, not run on hardware
//
#include <hip/hip_runtime.h>

#define H     1024
#define B0    64
#define FDIM  2048
#define ODIM  2048
#define NROW  8192
#define XLD   (H + B0 + FDIM)   /* 3136 */
#define NBLK  6

typedef __attribute__((ext_vector_type(16))) __bf16        v16bf;
typedef __attribute__((ext_vector_type(8)))  float         v8f;
typedef __attribute__((ext_vector_type(4)))  float         v4f;
typedef __attribute__((ext_vector_type(4)))  int           v4i;
typedef __attribute__((ext_vector_type(4)))  unsigned int  v4u;
typedef __attribute__((ext_vector_type(4)))  unsigned short v4us;

enum { EPI_NONE = 0, EPI_RELU = 1, EPI_ADD = 2, EPI_LEAKY_QSUB = 3 };

#if defined(__HIP_DEVICE_COMPILE__) && __has_builtin(__builtin_amdgcn_global_load_async_to_lds_b128) && __has_builtin(__builtin_amdgcn_s_wait_asynccnt)
#define USE_ASYNC 1
#define GLOBAL_AS __attribute__((address_space(1)))
#define LDS_AS    __attribute__((address_space(3)))
#else
#define USE_ASYNC 0
#endif

static __device__ __forceinline__ unsigned short f2bf(float f) {
  unsigned int u = __float_as_uint(f);
  unsigned int r = u + 0x7FFFu + ((u >> 16) & 1u);   // round-to-nearest-even
  return (unsigned short)(r >> 16);
}

// out[M,Nout] = epi( A_bf16[:, acol : acol+K] @ W_bf16[Nout,K]^T + bias )
// Both operands are bf16 in memory; tiles staged to LDS with async copies
// (double-buffered, ASYNCcnt) and consumed by v_wmma_f32_16x16x32_bf16.
// Tiling: block 128x128, BK=32; 8 waves, each wave = 64x32 (4x2 WMMA tiles).
__global__ __launch_bounds__(256)
void gemm_wmma_bf16(const unsigned short* __restrict__ A, int lda, int acol,
                    const unsigned short* __restrict__ W,
                    int K, int Nout,
                    const float* __restrict__ bias,
                    float* __restrict__ out,
                    unsigned short* __restrict__ out16,   // optional bf16 mirror
                    int epi,
                    const float* __restrict__ extra,      // ADD: addend, LEAKY_QSUB: minuend (q)
                    const float* __restrict__ alphap)     // LEAKY_QSUB: leaky slope
{
  __shared__ unsigned short As[2][128][32];
  __shared__ unsigned short Bs[2][128][32];

  const int tid  = threadIdx.x;
  const int lane = tid & 31;
  const int wave = tid >> 5;           // 0..7
  const int wm   = (wave >> 2) * 64;   // 0 / 64
  const int wn   = (wave & 3)  * 32;   // 0..96
  const int bm   = blockIdx.x * 128;
  const int bn   = blockIdx.y * 128;
  const int lr   = lane & 15;
  const int lh   = lane >> 4;

  v8f zero = {0.f, 0.f, 0.f, 0.f, 0.f, 0.f, 0.f, 0.f};
  v8f acc[4][2];
  #pragma unroll
  for (int i = 0; i < 4; ++i)
    #pragma unroll
    for (int j = 0; j < 2; ++j)
      acc[i][j] = zero;

  // Stage one 128x32 bf16 tile pair into LDS buffer `buf` for K-step `kt`.
  // 512 16B-chunks per tile / 256 threads = 2 chunks each (A and B).
  auto stage = [&](int buf, int kt) {
    #pragma unroll
    for (int c = 0; c < 2; ++c) {
      int ci  = c * 256 + tid;
      int row = ci >> 2;               // 4 x 16B chunks per 32-wide bf16 row
      int col = (ci & 3) * 8;
      const unsigned short* ga = &A[(size_t)(bm + row) * lda + (acol + kt * 32 + col)];
      const unsigned short* gb = &W[(size_t)(bn + row) * K + (kt * 32 + col)];
#if USE_ASYNC
      __builtin_amdgcn_global_load_async_to_lds_b128(
          (GLOBAL_AS v4i*)ga, (LDS_AS v4i*)&As[buf][row][col], 0, 0);
      __builtin_amdgcn_global_load_async_to_lds_b128(
          (GLOBAL_AS v4i*)gb, (LDS_AS v4i*)&Bs[buf][row][col], 0, 0);
#else
      *(v4u*)&As[buf][row][col] = *(const v4u*)ga;
      *(v4u*)&Bs[buf][row][col] = *(const v4u*)gb;
#endif
    }
  };

  const int KT = K >> 5;
  stage(0, 0);

  for (int kt = 0; kt < KT; ++kt) {
    const int  cur  = kt & 1;
    const bool more = (kt + 1 < KT);
    if (more) stage(cur ^ 1, kt + 1);   // prefetch next tile into other buffer
#if USE_ASYNC
    if (more) __builtin_amdgcn_s_wait_asynccnt(4);   // current tile's 4 copies done
    else      __builtin_amdgcn_s_wait_asynccnt(0);
#endif
    __syncthreads();                     // all waves' copies visible

    v16bf afrag[4], bfrag[2];
    #pragma unroll
    for (int i = 0; i < 4; ++i)
      afrag[i] = *(const v16bf*)&As[cur][wm + i * 16 + lr][lh * 16];
    #pragma unroll
    for (int j = 0; j < 2; ++j)
      bfrag[j] = *(const v16bf*)&Bs[cur][wn + j * 16 + lr][lh * 16];

    #pragma unroll
    for (int i = 0; i < 4; ++i)
      #pragma unroll
      for (int j = 0; j < 2; ++j)
        acc[i][j] = __builtin_amdgcn_wmma_f32_16x16x32_bf16(
            false, afrag[i], false, bfrag[j], (short)0, acc[i][j], false, false);

    __syncthreads();                     // done reading buf[cur] before overwrite
  }

  const float a_lk = (epi == EPI_LEAKY_QSUB) ? alphap[0] : 0.f;

  // C/D layout: VGPR v -> M = v (lanes 0-15) or v+8 (lanes 16-31); N = lane&15.
  #pragma unroll
  for (int i = 0; i < 4; ++i) {
    #pragma unroll
    for (int j = 0; j < 2; ++j) {
      const int col = bn + wn + j * 16 + lr;
      const float bv = bias[col];
      #pragma unroll
      for (int v = 0; v < 8; ++v) {
        const int row = bm + wm + i * 16 + v + lh * 8;
        const size_t idx = (size_t)row * Nout + col;
        float val = acc[i][j][v] + bv;
        if (epi == EPI_RELU) {
          val = fmaxf(val, 0.f);
        } else if (epi == EPI_ADD) {
          val = val + extra[idx];
        } else if (epi == EPI_LEAKY_QSUB) {
          float z = (val >= 0.f) ? val : a_lk * val;
          val = extra[idx] - z;
        }
        out[idx] = val;
        if (out16) out16[idx] = f2bf(val);
      }
    }
  }
}

// ---- one-shot f32 -> bf16 conversion (weights, x) ----
__global__ void cvt_f32_bf16(const float* __restrict__ s,
                             unsigned short* __restrict__ d, size_t n4) {
  size_t i = (size_t)blockIdx.x * blockDim.x + threadIdx.x;
  if (i >= n4) return;
  v4f v = *(const v4f*)&s[i * 4];
  v4us o = { f2bf(v.x), f2bf(v.y), f2bf(v.z), f2bf(v.w) };
  *(v4us*)&d[i * 4] = o;
}

// ---- elementwise glue (write f32 master + bf16 mirror for GEMM inputs) ----
__global__ void ew_square_from_x(const float* __restrict__ x, float* __restrict__ q,
                                 unsigned short* __restrict__ q16, size_t total) {
  size_t i = (size_t)blockIdx.x * blockDim.x + threadIdx.x;
  if (i >= total) return;
  size_t r = i / FDIM, c = i - r * FDIM;
  float d = x[r * XLD + (H + B0) + c];
  float v = 0.78539816339744830961f * d * d;   // pi/4 * d^2
  q[i] = v;
  q16[i] = f2bf(v);
}

__global__ void ew_mul(const float* __restrict__ a, const float* __restrict__ b,
                       float* __restrict__ o, unsigned short* __restrict__ o16,
                       size_t total) {
  size_t i = (size_t)blockIdx.x * blockDim.x + threadIdx.x;
  if (i >= total) return;
  float v = a[i] * b[i];
  o[i] = v;
  o16[i] = f2bf(v);
}

__global__ void ew_hid(const float* __restrict__ Aq, const float* __restrict__ q,
                       const float* __restrict__ sq, float* __restrict__ o,
                       unsigned short* __restrict__ o16, size_t total) {
  size_t i = (size_t)blockIdx.x * blockDim.x + threadIdx.x;
  if (i >= total) return;
  float v = Aq[i] * (q[i] + sq[i]);
  o[i] = v;
  o16[i] = f2bf(v);
}

__global__ void ew_tmp2(const float* __restrict__ h, const float* __restrict__ sh,
                        const float* __restrict__ rqh, const float* __restrict__ Dh,
                        float* __restrict__ o, unsigned short* __restrict__ o16,
                        size_t total) {
  size_t i = (size_t)blockIdx.x * blockDim.x + threadIdx.x;
  if (i >= total) return;
  float v = (h[i] + sh[i] + rqh[i]) * Dh[i];
  o[i] = v;
  o16[i] = f2bf(v);
}

extern "C" void kernel_launch(void* const* d_in, const int* in_sizes, int n_in,
                              void* d_out, int out_size, void* d_ws, size_t ws_size,
                              hipStream_t stream) {
  const float* x       = (const float*)d_in[0];
  const float* W_h0q   = (const float*)d_in[1];
  const float* b_h0q   = (const float*)d_in[2];
  const float* W_sq    = (const float*)d_in[3];
  const float* b_sq    = (const float*)d_in[4];
  const float* W_h0h   = (const float*)d_in[5];
  const float* b_h0h   = (const float*)d_in[6];
  const float* W_S     = (const float*)d_in[7];
  const float* b_S     = (const float*)d_in[8];
  const float* W_q0h   = (const float*)d_in[9];
  const float* b_q0h   = (const float*)d_in[10];
  const float* Wb_Aq   = (const float*)d_in[11];
  const float* bb_Aq   = (const float*)d_in[12];
  const float* Wb_Dh   = (const float*)d_in[13];
  const float* bb_Dh   = (const float*)d_in[14];
  const float* Wb_fh   = (const float*)d_in[15];
  const float* bb_fh   = (const float*)d_in[16];
  const float* Wb_hf   = (const float*)d_in[17];
  const float* bb_hf   = (const float*)d_in[18];
  const float* a_hf    = (const float*)d_in[19];
  const float* Wb_resq = (const float*)d_in[20];
  const float* bb_resq = (const float*)d_in[21];
  const float* W_out   = (const float*)d_in[22];
  const float* b_out   = (const float*)d_in[23];
  float* outp = (float*)d_out;

  const size_t NF = (size_t)NROW * FDIM;
  const size_t NH = (size_t)NROW * H;
  const size_t FF = (size_t)FDIM * FDIM;
  const size_t HF = (size_t)H * FDIM;

  char* wsp = (char*)d_ws;
  auto carveF = [&](size_t elems) {
    float* p = (float*)wsp;
    wsp += (elems * 4 + 255) & ~(size_t)255;
    return p;
  };
  auto carveU = [&](size_t elems) {
    unsigned short* p = (unsigned short*)wsp;
    wsp += (elems * 2 + 255) & ~(size_t)255;
    return p;
  };

  // f32 masters
  float* qb   = carveF(NF);   // q
  float* sumq = carveF(NF);   // res_h0_q then sum_q_const
  float* sumh = carveF(NH);   // sum_h_const
  float* resS = carveF(NF);   // res_S_q
  float* rqh  = carveF(NH);   // res_q_h
  float* Aq   = carveF(NF);   // A_q
  float* Dh   = carveF(NH);   // D_h
  float* tF   = carveF(NF);   // tmp1 / hid
  float* hb   = carveF(NH);   // h
  float* tH   = carveF(NH);   // tmp2
  // bf16 mirrors of GEMM inputs
  unsigned short* qb16 = carveU(NF);
  unsigned short* Aq16 = carveU(NF);
  unsigned short* tF16 = carveU(NF);
  unsigned short* tH16 = carveU(NH);
  unsigned short* x16  = carveU((size_t)NROW * XLD);
  // bf16 weights
  unsigned short* w_h0q16 = carveU((size_t)FDIM * B0);
  unsigned short* w_sq16  = carveU((size_t)FDIM * H);
  unsigned short* w_h0h16 = carveU((size_t)H * B0);
  unsigned short* w_S16   = carveU(FF);
  unsigned short* w_q0h16 = carveU(HF);
  unsigned short* w_Aq16  = carveU(NBLK * FF);
  unsigned short* w_Dh16  = carveU(NBLK * HF);
  unsigned short* w_fh16  = carveU(NBLK * HF);
  unsigned short* w_hf16  = carveU(NBLK * HF);
  unsigned short* w_rq16  = carveU(NBLK * HF);
  unsigned short* w_out16 = carveU((size_t)ODIM * FDIM);

  const int EWB = 256;
  auto cvt = [&](const float* s, unsigned short* d, size_t n) {
    size_t n4 = n / 4;
    hipLaunchKernelGGL(cvt_f32_bf16, dim3((unsigned)((n4 + EWB - 1) / EWB)),
                       dim3(EWB), 0, stream, s, d, n4);
  };
  cvt(x,       x16,     (size_t)NROW * XLD);
  cvt(W_h0q,   w_h0q16, (size_t)FDIM * B0);
  cvt(W_sq,    w_sq16,  (size_t)FDIM * H);
  cvt(W_h0h,   w_h0h16, (size_t)H * B0);
  cvt(W_S,     w_S16,   FF);
  cvt(W_q0h,   w_q0h16, HF);
  cvt(Wb_Aq,   w_Aq16,  NBLK * FF);
  cvt(Wb_Dh,   w_Dh16,  NBLK * HF);
  cvt(Wb_fh,   w_fh16,  NBLK * HF);
  cvt(Wb_hf,   w_hf16,  NBLK * HF);
  cvt(Wb_resq, w_rq16,  NBLK * HF);
  cvt(W_out,   w_out16, (size_t)ODIM * FDIM);

  const dim3 blk(256);
  auto gemm = [&](const unsigned short* A, int lda, int acol,
                  const unsigned short* W, int K, int Nout,
                  const float* bias, float* o, unsigned short* o16,
                  int epi, const float* extra, const float* alpha) {
    dim3 grid(NROW / 128, Nout / 128);
    hipLaunchKernelGGL(gemm_wmma_bf16, grid, blk, 0, stream,
                       A, lda, acol, W, K, Nout, bias, o, o16, epi, extra, alpha);
  };
  const unsigned gNF = (unsigned)((NF + EWB - 1) / EWB);
  const unsigned gNH = (unsigned)((NH + EWB - 1) / EWB);

  // ---- pre-stage ----
  gemm(x16, XLD, H,      w_h0q16, B0,   FDIM, b_h0q, sumq, nullptr, EPI_NONE, nullptr, nullptr);
  gemm(x16, XLD, 0,      w_sq16,  H,    FDIM, b_sq,  sumq, nullptr, EPI_ADD,  sumq,    nullptr);
  gemm(x16, XLD, H,      w_h0h16, B0,   H,    b_h0h, sumh, nullptr, EPI_NONE, nullptr, nullptr);
  gemm(x16, XLD, H + B0, w_S16,   FDIM, FDIM, b_S,   resS, nullptr, EPI_RELU, nullptr, nullptr);
  hipLaunchKernelGGL(ew_square_from_x, dim3(gNF), dim3(EWB), 0, stream, x, qb, qb16, NF);
  gemm(qb16, FDIM, 0,    w_q0h16, FDIM, H,    b_q0h, rqh,  nullptr, EPI_NONE, nullptr, nullptr);

  // ---- 6 blocks ----
  for (int i = 0; i < NBLK; ++i) {
    const unsigned short* W_Aq_i   = w_Aq16 + (size_t)i * FF;
    const unsigned short* W_Dh_i   = w_Dh16 + (size_t)i * HF;
    const unsigned short* W_fh_i   = w_fh16 + (size_t)i * HF;
    const unsigned short* W_hf_i   = w_hf16 + (size_t)i * HF;
    const unsigned short* W_rq_i   = w_rq16 + (size_t)i * HF;
    const float* b_Aq_i   = bb_Aq   + (size_t)i * FDIM;
    const float* b_Dh_i   = bb_Dh   + (size_t)i * H;
    const float* b_fh_i   = bb_fh   + (size_t)i * H;
    const float* b_hf_i   = bb_hf   + (size_t)i * FDIM;
    const float* b_resq_i = bb_resq + (size_t)i * H;

    hipLaunchKernelGGL(ew_mul, dim3(gNF), dim3(EWB), 0, stream, qb, resS, tF, tF16, NF);
    gemm(tF16, FDIM, 0, W_Aq_i, FDIM, FDIM, b_Aq_i, Aq, Aq16, EPI_NONE, nullptr, nullptr);
    gemm(Aq16, FDIM, 0, W_Dh_i, FDIM, H,    b_Dh_i, Dh, nullptr, EPI_RELU, nullptr, nullptr);
    hipLaunchKernelGGL(ew_hid, dim3(gNF), dim3(EWB), 0, stream, Aq, qb, sumq, tF, tF16, NF);
    gemm(tF16, FDIM, 0, W_fh_i, FDIM, H,    b_fh_i, hb, nullptr, EPI_RELU, nullptr, nullptr);
    hipLaunchKernelGGL(ew_tmp2, dim3(gNH), dim3(EWB), 0, stream, hb, sumh, rqh, Dh, tH, tH16, NH);
    gemm(tH16, H, 0, W_hf_i, H, FDIM, b_hf_i, qb, qb16, EPI_LEAKY_QSUB, qb, a_hf + i);
    gemm(qb16, FDIM, 0, W_rq_i, FDIM, H, b_resq_i, rqh, nullptr, EPI_RELU, nullptr, nullptr);
  }

  // ---- output ----
  gemm(qb16, FDIM, 0, w_out16, FDIM, ODIM, b_out, outp, nullptr, EPI_NONE, nullptr, nullptr);
}
